// GRU_D_87832081203990
// MI455X (gfx1250) — compile-verified, matrix operand order checked by reading, not run on hardware
//
#include <hip/hip_runtime.h>
#include <hip/hip_bf16.h>

// Problem constants (from reference)
static constexpr int STEPS = 128;
static constexpr int NBAT  = 1024;
static constexpr int HD    = 512;
static constexpr int CMB   = 514;   // H + 2
static constexpr int KPAD  = 544;   // 17 * 32 (K padded for bf16 WMMA k=32)
static constexpr int KCH   = 17;    // K chunks of 32
static constexpr int ROWS  = 32;    // batch rows per workgroup (two M tiles)

// packed-weight strides (in ushorts)
static constexpr int FRAG   = 512;           // one 32-lane x 16-bf16 fragment
static constexpr int NTSTR  = KCH * FRAG;    // per-ntile stride    (8704  -> 17408 B)
static constexpr int GATER  = 32 * NTSTR;    // z->r gate offset    (278528 -> 557056 B)

typedef __attribute__((ext_vector_type(16))) __bf16 v16bf;
typedef __attribute__((ext_vector_type(8)))  float  v8f;

static __device__ __forceinline__ unsigned short f2bf(float f) {
  unsigned int u = __float_as_uint(f);
  u += 0x7FFFu + ((u >> 16) & 1u);            // round-to-nearest-even
  return (unsigned short)(u >> 16);
}

static __device__ __forceinline__ float fast_sigmoid(float x) {
  return __fdividef(1.0f, 1.0f + __expf(-x));
}
static __device__ __forceinline__ float fast_tanh(float x) {
  return 1.0f - 2.0f * __fdividef(1.0f, __expf(2.0f * x) + 1.0f);
}

static __device__ __forceinline__ v16bf ld_frag16(const unsigned short* p0,
                                                  const unsigned short* p1) {
  v16bf a;
  ((uint4*)&a)[0] = *(const uint4*)p0;        // 8 bf16 (K run +0..7)
  ((uint4*)&a)[1] = *(const uint4*)p1;        // 8 bf16 (K run +16..23)
  return a;
}

#define WMMA_BF16(A, B, C) \
  __builtin_amdgcn_wmma_f32_16x16x32_bf16(false, (A), false, (B), (short)0, (C), false, false)

// ---------------------------------------------------------------------------
// Pack Wz|Wr (g_nt 0..63) and Wh (g_nt 64..95) into per-lane WMMA B-fragment
// order. Comb K layout is PERMUTED to [s(0..511), x, m, pad]:
//   k<512 -> W column k+1 ; k==512 -> W column 0 (x) ; k==513 -> W column 513.
// Fragment(g_nt,kc) = 32 lanes x 16 bf16; lane = (hl<<4)|n;
// element e=2i+h -> K = (i<4?0:16) + hl*8 + 2*(i&3) + h (mirrors A layout).
// ---------------------------------------------------------------------------
__global__ void pack_w(const float* __restrict__ Wz, const float* __restrict__ Wr,
                       const float* __restrict__ Wh,
                       unsigned short* __restrict__ pzr,
                       unsigned short* __restrict__ ph) {
  int tid = blockIdx.x * blockDim.x + threadIdx.x;
  const int total = 96 * KCH * 32;
  if (tid >= total) return;
  int lane = tid & 31;
  int rest = tid >> 5;
  int kc   = rest % KCH;
  int gnt  = rest / KCH;                       // 0..95
  const float* W = (gnt < 32) ? Wz : ((gnt < 64) ? Wr : Wh);
  unsigned short* dst = (gnt < 64)
      ? (pzr + ((size_t)(gnt * KCH + kc) * 32 + lane) * 16)
      : (ph  + ((size_t)((gnt - 64) * KCH + kc) * 32 + lane) * 16);
  int n   = lane & 15;
  int hlh = lane >> 4;
  int j   = (gnt & 31) * 16 + n;               // output column -> W row j
#pragma unroll
  for (int e = 0; e < 16; ++e) {
    int i  = e >> 1, hh = e & 1;
    int kl = ((i < 4) ? 0 : 16) + hlh * 8 + 2 * (i & 3) + hh;
    int k  = kc * 32 + kl;
    int kcol = (k < HD) ? (k + 1) : ((k == HD) ? 0 : ((k == HD + 1) ? (CMB - 1) : -1));
    float v = (kcol >= 0) ? W[(size_t)j * CMB + kcol] : 0.0f;
    dst[e] = f2bf(v);
  }
}

// ---------------------------------------------------------------------------
// Fully fused GRU-D: one WG owns 32 batch rows (2 M-tiles) for all 128 steps.
// Each B fragment is reused by 2 WMMAs; kc=0 B fragments of every GEMM are
// preloaded before the preceding barrier so no GEMM starts cold on L2.
// ---------------------------------------------------------------------------
__global__ __launch_bounds__(256) void gru_d_fused(
    const float* __restrict__ inp, const float* __restrict__ hidden,
    const float* __restrict__ Wg, const float* __restrict__ bg,
    const float* __restrict__ bz, const float* __restrict__ br,
    const float* __restrict__ bh, const float* __restrict__ Wo,
    const float* __restrict__ bo, const float* __restrict__ meanp,
    const unsigned short* __restrict__ pzr, const unsigned short* __restrict__ ph,
    float* __restrict__ out) {
  extern __shared__ char smem[];
  float*          sh_h  = (float*)smem;                        // ROWS*HD f32
  unsigned short* sh_c1 = (unsigned short*)(sh_h + ROWS * HD); // ROWS*KPAD bf16
  unsigned short* sh_c2 = sh_c1 + ROWS * KPAD;                 // ROWS*KPAD bf16
  float* sh_bz = (float*)(sh_c2 + ROWS * KPAD);
  float* sh_br = sh_bz + HD;
  float* sh_bh = sh_br + HD;
  float* sh_g  = sh_bh + HD;                                   // ROWS
  float* sh_xp = sh_g + ROWS;                                  // ROWS

  const int tid  = threadIdx.x;
  const int lane = tid & 31;
  const int wave = tid >> 5;
  const int hl   = lane >> 4;
  const int nn   = lane & 15;
  const int blk  = blockIdx.x;

  const float Wg0   = Wg[0];
  const float bg0   = bg[0];
  const float meanv = meanp[0];
  const v8f zero = {0.f, 0.f, 0.f, 0.f, 0.f, 0.f, 0.f, 0.f};

  // per-wave packed-weight base pointers (all tile/gate offsets are immediates)
  const unsigned short* pwz = pzr + (size_t)(wave * 4) * NTSTR + (size_t)lane * 16;
  const unsigned short* pwh = ph  + (size_t)(wave * 4) * NTSTR + (size_t)lane * 16;

  // ---- init: h0, biases, xprime=0, zero K padding ----
  for (int idx = tid; idx < ROWS * HD; idx += 256) {
    int r = idx >> 9, j = idx & (HD - 1);
    sh_h[idx] = hidden[(size_t)(blk * ROWS + r) * HD + j];
  }
  for (int j = tid; j < HD; j += 256) { sh_bz[j] = bz[j]; sh_br[j] = br[j]; sh_bh[j] = bh[j]; }
  if (tid < ROWS) sh_xp[tid] = 0.0f;
  for (int idx = tid; idx < ROWS * (KPAD - CMB); idx += 256) {
    int r = idx / (KPAD - CMB), c = idx % (KPAD - CMB);
    sh_c1[r * KPAD + CMB + c] = 0;
    sh_c2[r * KPAD + CMB + c] = 0;
  }

  // warm preload: kc=0 fragments for the first z/r GEMM
  v16bf pzb[4], prb[4];
#pragma unroll
  for (int tt = 0; tt < 4; ++tt) {
    pzb[tt] = ld_frag16(pwz + tt * NTSTR,         pwz + tt * NTSTR + 8);
    prb[tt] = ld_frag16(pwz + GATER + tt * NTSTR, pwz + GATER + tt * NTSTR + 8);
  }
  __syncthreads();

  for (int t = 0; t < STEPS; ++t) {
    // ---- phase 1a: per-row scalars (gamma, imputation) ----
    if (tid < ROWS) {
      const int r = tid;
      const float* ip = inp + ((size_t)t * NBAT + blk * ROWS + r) * 3;
      float x = ip[0], mv = ip[1], dtv = ip[2];
      float g = __expf(-fmaxf(dtv * Wg0 + bg0, 0.0f));
      float xp = sh_xp[r];
      if (x != x) {                                   // isnan -> impute
        x = (1.0f - mv) * (g * xp + meanv * (1.0f - g));
      } else {
        xp = x;                                       // observed -> update x'
      }
      sh_xp[r] = xp;
      sh_g[r]  = g;
      unsigned short xb = f2bf(x), mb = f2bf(mv);
      sh_c1[r * KPAD + HD] = xb;  sh_c1[r * KPAD + HD + 1] = mb;
      sh_c2[r * KPAD + HD] = xb;  sh_c2[r * KPAD + HD + 1] = mb;
    }
    __syncthreads();

    // ---- phase 1b: comb s-part = gamma*h (two bf16 per 32-bit DS store) ----
    for (int idx = tid; idx < ROWS * (HD / 2); idx += 256) {
      int r  = idx >> 8;                 // HD/2 = 256 pairs per row
      int j2 = (idx & 255) * 2;
      float g  = sh_g[r];
      float h0 = sh_h[r * HD + j2];
      float h1 = sh_h[r * HD + j2 + 1];
      unsigned int pk = (unsigned int)f2bf(g * h0) | ((unsigned int)f2bf(g * h1) << 16);
      *(unsigned int*)(sh_c1 + r * KPAD + j2) = pk;
    }
    __syncthreads();

    // ---- phase 2: z & r GEMMs (kc=0 uses preloaded B; 16-trip main loop) ----
    v8f accz[2][4], accr[2][4];
#pragma unroll
    for (int mt = 0; mt < 2; ++mt)
#pragma unroll
      for (int tt = 0; tt < 4; ++tt) { accz[mt][tt] = zero; accr[mt][tt] = zero; }

    const unsigned short* ar0 = sh_c1 + nn * KPAD + hl * 8;
    const unsigned short* ar1 = ar0 + 16 * KPAD;
    {
      v16bf a0 = ld_frag16(ar0, ar0 + 16);
      v16bf a1 = ld_frag16(ar1, ar1 + 16);
#pragma unroll
      for (int tt = 0; tt < 4; ++tt) {
        accz[0][tt] = WMMA_BF16(a0, pzb[tt], accz[0][tt]);
        accz[1][tt] = WMMA_BF16(a1, pzb[tt], accz[1][tt]);
        accr[0][tt] = WMMA_BF16(a0, prb[tt], accr[0][tt]);
        accr[1][tt] = WMMA_BF16(a1, prb[tt], accr[1][tt]);
      }
    }
    {
      const unsigned short* p = pwz + FRAG;     // kc = 1
      const unsigned short* a0p = ar0 + 32;
      const unsigned short* a1p = ar1 + 32;
#pragma unroll 2
      for (int kc = 1; kc < KCH; ++kc) {
        v16bf a0 = ld_frag16(a0p, a0p + 16);
        v16bf a1 = ld_frag16(a1p, a1p + 16);
#pragma unroll
        for (int tt = 0; tt < 4; ++tt) {
          v16bf bz0 = ld_frag16(p + tt * NTSTR, p + tt * NTSTR + 8);
          accz[0][tt] = WMMA_BF16(a0, bz0, accz[0][tt]);
          accz[1][tt] = WMMA_BF16(a1, bz0, accz[1][tt]);
          v16bf br0 = ld_frag16(p + GATER + tt * NTSTR, p + GATER + tt * NTSTR + 8);
          accr[0][tt] = WMMA_BF16(a0, br0, accr[0][tt]);
          accr[1][tt] = WMMA_BF16(a1, br0, accr[1][tt]);
        }
        p += FRAG; a0p += 32; a1p += 32;
      }
    }

    // ---- phase 3: preload Wh kc=0; sigmoids; z in regs, r*s -> comb2 ----
    v16bf phb[4];
#pragma unroll
    for (int tt = 0; tt < 4; ++tt)
      phb[tt] = ld_frag16(pwh + tt * NTSTR, pwh + tt * NTSTR + 8);

#pragma unroll
    for (int mt = 0; mt < 2; ++mt)
#pragma unroll
      for (int tt = 0; tt < 4; ++tt) {
        const int j0 = (wave * 4 + tt) * 16 + nn;
#pragma unroll
        for (int i = 0; i < 8; ++i) {
          const int me = mt * 16 + i + 8 * hl;
          float zv = fast_sigmoid(accz[mt][tt][i] + sh_bz[j0]);
          accz[mt][tt][i] = zv;
          float rv = fast_sigmoid(accr[mt][tt][i] + sh_br[j0]);
          float sv = sh_g[me] * sh_h[me * HD + j0];
          sh_c2[me * KPAD + j0] = f2bf(rv * sv);
        }
      }
    __syncthreads();

    // ---- phase 4: h_tilde GEMM (kc=0 preloaded) ----
    v8f acch[2][4];
#pragma unroll
    for (int mt = 0; mt < 2; ++mt)
#pragma unroll
      for (int tt = 0; tt < 4; ++tt) acch[mt][tt] = zero;

    const unsigned short* br0p = sh_c2 + nn * KPAD + hl * 8;
    const unsigned short* br1p = br0p + 16 * KPAD;
    {
      v16bf a0 = ld_frag16(br0p, br0p + 16);
      v16bf a1 = ld_frag16(br1p, br1p + 16);
#pragma unroll
      for (int tt = 0; tt < 4; ++tt) {
        acch[0][tt] = WMMA_BF16(a0, phb[tt], acch[0][tt]);
        acch[1][tt] = WMMA_BF16(a1, phb[tt], acch[1][tt]);
      }
    }
    {
      const unsigned short* p = pwh + FRAG;     // kc = 1
      const unsigned short* a0p = br0p + 32;
      const unsigned short* a1p = br1p + 32;
#pragma unroll 2
      for (int kc = 1; kc < KCH; ++kc) {
        v16bf a0 = ld_frag16(a0p, a0p + 16);
        v16bf a1 = ld_frag16(a1p, a1p + 16);
#pragma unroll
        for (int tt = 0; tt < 4; ++tt) {
          v16bf b0 = ld_frag16(p + tt * NTSTR, p + tt * NTSTR + 8);
          acch[0][tt] = WMMA_BF16(a0, b0, acch[0][tt]);
          acch[1][tt] = WMMA_BF16(a1, b0, acch[1][tt]);
        }
        p += FRAG; a0p += 32; a1p += 32;
      }
    }

    // ---- phase 5: preload next step's z/r kc=0; gate update of h ----
#pragma unroll
    for (int tt = 0; tt < 4; ++tt) {
      pzb[tt] = ld_frag16(pwz + tt * NTSTR,         pwz + tt * NTSTR + 8);
      prb[tt] = ld_frag16(pwz + GATER + tt * NTSTR, pwz + GATER + tt * NTSTR + 8);
    }
#pragma unroll
    for (int mt = 0; mt < 2; ++mt)
#pragma unroll
      for (int tt = 0; tt < 4; ++tt) {
        const int j0 = (wave * 4 + tt) * 16 + nn;
#pragma unroll
        for (int i = 0; i < 8; ++i) {
          const int me = mt * 16 + i + 8 * hl;
          float htl = fast_tanh(acch[mt][tt][i] + sh_bh[j0]);
          float sv  = sh_g[me] * sh_h[me * HD + j0];
          float zv  = accz[mt][tt][i];
          sh_h[me * HD + j0] = (1.0f - zv) * htl + zv * sv;
        }
      }
    __syncthreads();
  }

  // ---- epilogue: prediction = softmax(h @ Wo.T + bo), and hT ----
  if (tid < ROWS) {
    const int r  = tid;
    const int gn = blk * ROWS + r;
    float o0 = bo[0], o1 = bo[1];
    for (int j = 0; j < HD; ++j) {
      float hv = sh_h[r * HD + j];
      o0 += Wo[j] * hv;
      o1 += Wo[HD + j] * hv;
    }
    float mx = fmaxf(o0, o1);
    float e0 = __expf(o0 - mx), e1 = __expf(o1 - mx);
    float inv = __fdividef(1.0f, e0 + e1);
    out[gn * 2 + 0] = e0 * inv;
    out[gn * 2 + 1] = e1 * inv;
  }
  for (int idx = tid; idx < ROWS * HD; idx += 256) {
    int r = idx >> 9, j = idx & (HD - 1);
    out[2 * NBAT + (size_t)(blk * ROWS + r) * HD + j] = sh_h[idx];
  }
}

extern "C" void kernel_launch(void* const* d_in, const int* in_sizes, int n_in,
                              void* d_out, int out_size, void* d_ws, size_t ws_size,
                              hipStream_t stream) {
  const float* inp    = (const float*)d_in[0];
  const float* hidden = (const float*)d_in[1];
  const float* Wg     = (const float*)d_in[2];
  const float* bg     = (const float*)d_in[3];
  const float* Wz     = (const float*)d_in[4];
  const float* bz     = (const float*)d_in[5];
  const float* Wr     = (const float*)d_in[6];
  const float* br     = (const float*)d_in[7];
  const float* Wh     = (const float*)d_in[8];
  const float* bh     = (const float*)d_in[9];
  const float* Wo     = (const float*)d_in[10];
  const float* bo     = (const float*)d_in[11];
  const float* meanp  = (const float*)d_in[12];

  unsigned short* pzr = (unsigned short*)d_ws;                 // 64*17*32*16 bf16
  unsigned short* ph  = pzr + (size_t)64 * KCH * 32 * 16;      // 32*17*32*16 bf16

  const int packThreads = 96 * KCH * 32;
  pack_w<<<(packThreads + 255) / 256, 256, 0, stream>>>(Wz, Wr, Wh, pzr, ph);

  const size_t smem = (size_t)ROWS * HD * 4        // h (f32)
                    + 2 * (size_t)ROWS * KPAD * 2  // comb, comb2 (bf16)
                    + 3 * (size_t)HD * 4           // biases
                    + 2 * (size_t)ROWS * 4;        // gamma, xprime
  gru_d_fused<<<NBAT / ROWS, 256, smem, stream>>>(
      inp, hidden, Wg, bg, bz, br, bh, Wo, bo, meanp, pzr, ph, (float*)d_out);
}
